// Region_Proposal_Network_11622181503318
// MI455X (gfx1250) — matrix operand order repaired and checked by reading, not compiled, not run on hardware
//
#include <hip/hip_runtime.h>

// ---------------------------------------------------------------------------
// Types for CDNA5 WMMA / TDM
// ---------------------------------------------------------------------------
typedef _Float16 h16_t;
typedef _Float16 v16h __attribute__((ext_vector_type(16)));
typedef _Float16 v8h  __attribute__((ext_vector_type(8)));
typedef float    v8f  __attribute__((ext_vector_type(8)));
typedef unsigned u32x4 __attribute__((ext_vector_type(4)));
typedef int      i32x4 __attribute__((ext_vector_type(4)));
typedef int      i32x8 __attribute__((ext_vector_type(8)));

union AFrag { v16h v; v8h h[2]; _Float16 e[16]; };
union CFrag { v8f  v; float e[8]; };

// ---------------------------------------------------------------------------
// Problem constants
// ---------------------------------------------------------------------------
#define HDIM 128
#define WDIM 128
#define CIN  512
#define COUT 512
#define NPOS (HDIM * WDIM)          // 16384 spatial positions
#define KANCH 9
#define NANCH (NPOS * KANCH)        // 147456 anchors
#define NUM_PRE 12000
#define NUM_POST 2000
#define KSTEPS 144                  // 9 taps x 16 channel-blocks of 32

// d_out regions (element offsets)
#define R0_LOCS   0                 // (147456,4)
#define R1_CLS    589824            // (147456,2)
#define R2_BOXES  884736            // (2000,4)
#define R3_IDX    892736            // (2000,) int32
#define R4_ANCH   894736            // (147456,4)
#define R5_MASK   1484560           // (2000,)

// workspace byte offsets
static constexpr size_t OFF_XT = 0;                                       // x f16 NHWC [pos][c]
static constexpr size_t OFF_WF = OFF_XT + (size_t)CIN * NPOS * 2;         // conv w f16, WMMA frag layout
static constexpr size_t OFF_HF = OFF_WF + (size_t)9 * CIN * COUT * 2;     // head w f16, WMMA frag layout
static constexpr size_t OFF_HH = OFF_HF + (size_t)CIN * 64 * 2;           // conv out f16 [pos][cout]
static constexpr size_t OFF_HO = OFF_HH + (size_t)NPOS * COUT * 2;        // head out f32 [pos][64]
static constexpr size_t OFF_BX = OFF_HO + (size_t)NPOS * 64 * 4;          // decoded boxes f32 [NANCH][4]
static constexpr size_t OFF_SC = OFF_BX + (size_t)NANCH * 4 * 4;          // scores f32 [NANCH]
static constexpr size_t OFF_CT = OFF_SC + (size_t)NANCH * 4;              // ctrl
static constexpr size_t OFF_TS = OFF_CT + 64;                             // top scores f32 [12000]
static constexpr size_t OFF_TB = OFF_TS + (size_t)NUM_PRE * 4;            // top boxes f32 [12000][4]
static constexpr size_t OFF_KP = OFF_TB + (size_t)NUM_PRE * 16;           // keep int [2000]

__device__ __forceinline__ unsigned fkey(float f) {
    unsigned u = __float_as_uint(f);
    return (u & 0x80000000u) ? ~u : (u | 0x80000000u);
}

// ---------------------------------------------------------------------------
// TDM: 1-D contiguous tile, global -> LDS. D# built per CDNA5 ISA §8.
// data_size = 8 bytes; bytes must be a multiple of 8; addrs 8B-aligned.
// ---------------------------------------------------------------------------
__device__ __forceinline__ void tdm_load_1d(unsigned lds_off, const void* gaddr, unsigned bytes) {
    unsigned long long ga = (unsigned long long)(uintptr_t)gaddr;
    const unsigned elems = bytes >> 3;          // 8-byte units
    u32x4 g0;
    g0[0] = 1u;                                 // count=1, user descriptor
    g0[1] = lds_off;                            // lds_addr
    g0[2] = (unsigned)(ga & 0xFFFFFFFFu);       // global_addr[31:0]
    g0[3] = (unsigned)((ga >> 32) & 0x1FFFFFFu) | (2u << 30);  // global_addr[56:32] | type=2
    i32x8 g1;
    g1[0] = (int)(3u << 16);                    // data_size=3 (8B), no multicast
    g1[1] = (int)((elems & 0xFFFFu) << 16);     // tensor_dim0[15:0]
    g1[2] = (int)(((elems >> 16) & 0xFFFFu) | (1u << 16)); // tensor_dim0[31:16] | tensor_dim1=1
    g1[3] = (int)((elems & 0xFFFFu) << 16);     // tile_dim0 = elems
    g1[4] = 1;                                  // tile_dim1 = 1
    g1[5] = (int)elems;                         // tensor_dim0_stride[31:0]
    g1[6] = 0;
    g1[7] = 0;
    i32x4 z4 = {0, 0, 0, 0};
#if defined(__clang_major__) && __clang_major__ >= 23
    i32x8 z8 = {0, 0, 0, 0, 0, 0, 0, 0};
    __builtin_amdgcn_tensor_load_to_lds(g0, g1, z4, z4, z8, 0);
#else
    __builtin_amdgcn_tensor_load_to_lds(g0, g1, z4, z4, 0);
#endif
}

// ---------------------------------------------------------------------------
// K0: layout transforms.
//   xt[pos][c]   : NCHW -> NHWC f16 (A-fragments become K-contiguous)
//   whf          : conv weights pre-swizzled to per-lane WMMA B-fragment order
//                  [t][cb][ntg][lane][16]  with B.e[q] = B[K=(lane>>4)*16+q][N=ntg*16+(lane&15)]
//   hwf          : head weights (36 reg + 18 cls + 10 pad) same frag order [ks][ntg][lane][16]
// ---------------------------------------------------------------------------
__global__ void k_convert(const float* __restrict__ x,
                          const float* __restrict__ conv_w,
                          const float* __restrict__ reg_w,
                          const float* __restrict__ cls_w,
                          h16_t* __restrict__ xt,
                          h16_t* __restrict__ whf,
                          h16_t* __restrict__ hwf) {
    const size_t NX = (size_t)CIN * NPOS;
    const size_t NW = (size_t)9 * CIN * COUT;
    const size_t NH = (size_t)CIN * 64;
    const size_t stride = (size_t)gridDim.x * blockDim.x;
    const size_t t0 = (size_t)blockIdx.x * blockDim.x + threadIdx.x;

    for (size_t i = t0; i < NX; i += stride) {
        const size_t p = i >> 9, c = i & 511;
        xt[i] = (h16_t)x[c * NPOS + p];
    }

    for (size_t i = t0; i < NW; i += stride) {
        const int q    = (int)(i & 15);
        const int lane = (int)((i >> 4) & 31);
        const int ntg  = (int)((i >> 9) & 31);
        const int cb   = (int)((i >> 14) & 15);
        const int t    = (int)(i >> 18);
        const int c = cb * 32 + (lane >> 4) * 16 + q;
        const int o = ntg * 16 + (lane & 15);
        whf[i] = (h16_t)conv_w[((size_t)o * CIN + c) * 9 + t];
    }

    for (size_t i = t0; i < NH; i += stride) {
        const int q    = (int)(i & 15);
        const int lane = (int)((i >> 4) & 31);
        const int ntg  = (int)((i >> 9) & 3);
        const int ks   = (int)(i >> 11);
        const int k = ks * 32 + (lane >> 4) * 16 + q;
        const int n = ntg * 16 + (lane & 15);
        float v = 0.0f;
        if (n < 36)      v = reg_w[(size_t)n * CIN + k];
        else if (n < 54) v = cls_w[(size_t)(n - 36) * CIN + k];
        hwf[i] = (h16_t)v;
    }
}

// ---------------------------------------------------------------------------
// K1: 3x3 conv as implicit GEMM. Block = one image row (M=128) x 128 out-ch.
// 8 waves, each 32(M)x64(N) = 2x4 WMMA accumulators.
// Double-buffered TDM pipeline: wave0 issues the NEXT K-step's 8KB weight
// block into the alternate LDS buffer, then waits s_wait_tensorcnt(1) so only
// the OLDEST (current) transfer is guaranteed complete -> DMA overlaps WMMA.
// A-fragments load branch-free straight from NHWC global memory.
// ---------------------------------------------------------------------------
__global__ __launch_bounds__(256) void k_conv(const h16_t* __restrict__ xt,
                                              const h16_t* __restrict__ whf,
                                              const float* __restrict__ conv_b,
                                              h16_t* __restrict__ hh) {
    __shared__ __align__(16) _Float16 Bsh[2][8 * 512];   // 2 x 8KB weight buffers

    const int row  = blockIdx.x;
    const int by   = blockIdx.y;          // out-channel block (128 ch)
    const int tid  = threadIdx.x;
    const int lane = tid & 31;
    const int wv   = tid >> 5;            // 0..7
    const int wm   = wv & 3;              // M sub-block (32 rows)
    const int wn   = wv >> 2;             // N sub-block (64 cols)
    const int mrow = lane & 15;
    const int khalf = lane >> 4;

    CFrag acc[2][4];
#pragma unroll
    for (int a = 0; a < 2; ++a)
#pragma unroll
        for (int b = 0; b < 4; ++b)
#pragma unroll
            for (int q = 0; q < 8; ++q) acc[a][b].e[q] = 0.0f;

    v8h zro;
#pragma unroll
    for (int q = 0; q < 8; ++q) zro[q] = (_Float16)0.0f;

    const unsigned lds0 = (unsigned)(uintptr_t)(&Bsh[0][0]);
    const h16_t* wsrc = whf + (size_t)by * 8 * 512;   // +it*16384 halves per K-step

    // prologue: stage step 0 into buffer 0
    if (wv == 0) tdm_load_1d(lds0, wsrc, 8192);

    for (int it = 0; it < KSTEPS; ++it) {
        const int t  = it >> 4;           // tap 0..8
        const int c0 = (it & 15) * 32;    // channel block
        const int kw = t % 3;
        const int r  = row + (t / 3) - 1;
        const bool rok = (r >= 0) && (r < HDIM);
        const int rc = r < 0 ? 0 : (r > HDIM - 1 ? HDIM - 1 : r);

        if (wv == 0) {
            if (it < KSTEPS - 1) {
                // kick off next step's DMA into the other buffer, then wait
                // for the *oldest* transfer only (in-order completion).
                tdm_load_1d(lds0 + (unsigned)(((it + 1) & 1) * 8192),
                            wsrc + (size_t)(it + 1) * 16384, 8192);
                __builtin_amdgcn_s_wait_tensorcnt(1);
            } else {
                __builtin_amdgcn_s_wait_tensorcnt(0);
            }
        }
        __syncthreads();                  // current buffer visible to all waves

        // ---- A fragments: branch-free direct loads from NHWC global
        AFrag af[2];
#pragma unroll
        for (int mt = 0; mt < 2; ++mt) {
            const int m = wm * 32 + mt * 16 + mrow;
            const int col = m + kw - 1;
            const bool ok = rok && (col >= 0) && (col < WDIM);
            const int colc = col < 0 ? 0 : (col > WDIM - 1 ? WDIM - 1 : col);
            const h16_t* ap = xt + ((size_t)(rc * WDIM + colc)) * CIN + c0 + khalf * 16;
            af[mt].h[0] = ok ? *(const v8h*)ap : zro;
            af[mt].h[1] = ok ? *(const v8h*)(ap + 8) : zro;
            __builtin_prefetch(ap + 32, 0, 1);   // next channel block
        }
        // ---- B fragments: contiguous 32B per lane from current LDS buffer
        const _Float16* bbuf = &Bsh[it & 1][0];
        AFrag bf[4];
#pragma unroll
        for (int nt = 0; nt < 4; ++nt) {
            const int base = ((wn * 4 + nt) * 32 + lane) * 16;
            bf[nt].h[0] = *(const v8h*)&bbuf[base];
            bf[nt].h[1] = *(const v8h*)&bbuf[base + 8];
        }
#pragma unroll
        for (int mt = 0; mt < 2; ++mt)
#pragma unroll
            for (int nt = 0; nt < 4; ++nt)
                acc[mt][nt].v = __builtin_amdgcn_wmma_f32_16x16x32_f16(
                    false, af[mt].v, false, bf[nt].v,
                    (short)0, acc[mt][nt].v, false, false);

        __syncthreads();                  // readers done before this buffer is
                                          // overwritten by the issue 2 steps on
    }

    // ---- epilogue: +bias, ReLU, f16 store
#pragma unroll
    for (int mt = 0; mt < 2; ++mt) {
#pragma unroll
        for (int nt = 0; nt < 4; ++nt) {
            const int ncol = by * 128 + wn * 64 + nt * 16 + (lane & 15);
            const float bias = conv_b[ncol];
#pragma unroll
            for (int rr = 0; rr < 8; ++rr) {
                const int m = wm * 32 + mt * 16 + khalf * 8 + rr;
                float v = acc[mt][nt].e[rr] + bias;
                v = v > 0.0f ? v : 0.0f;
                hh[((size_t)(row * WDIM + m)) * COUT + ncol] = (h16_t)v;
            }
        }
    }
}

// ---------------------------------------------------------------------------
// K2: 1x1 heads: (16384 x 512) x (512 x 64). Block = 16 positions, 4 waves,
// one 16x16 N-tile each; pure b128 fragment loads, no LDS.
// ---------------------------------------------------------------------------
__global__ __launch_bounds__(128) void k_head(const h16_t* __restrict__ hh,
                                              const h16_t* __restrict__ hwf,
                                              const float* __restrict__ reg_b,
                                              const float* __restrict__ cls_b,
                                              float* __restrict__ ho) {
    const int p0   = blockIdx.x * 16;
    const int tid  = threadIdx.x;
    const int lane = tid & 31;
    const int wv   = tid >> 5;            // N tile
    const int mrow = lane & 15;
    const int khalf = lane >> 4;
    const int ncol = wv * 16 + (lane & 15);

    CFrag acc;
#pragma unroll
    for (int q = 0; q < 8; ++q) acc.e[q] = 0.0f;

    for (int ks = 0; ks < 16; ++ks) {
        AFrag a, b;
        const h16_t* ap = hh + (size_t)(p0 + mrow) * COUT + ks * 32 + khalf * 16;
        a.h[0] = *(const v8h*)ap;
        a.h[1] = *(const v8h*)(ap + 8);
        const h16_t* bp = hwf + ((size_t)(ks * 4 + wv) * 32 + lane) * 16;
        b.h[0] = *(const v8h*)bp;
        b.h[1] = *(const v8h*)(bp + 8);
        acc.v = __builtin_amdgcn_wmma_f32_16x16x32_f16(
            false, a.v, false, b.v, (short)0, acc.v, false, false);
    }

    float bias = 0.0f;
    if (ncol < 36)      bias = reg_b[ncol];
    else if (ncol < 54) bias = cls_b[ncol - 36];
#pragma unroll
    for (int rr = 0; rr < 8; ++rr) {
        const int m = khalf * 8 + rr;
        ho[(size_t)(p0 + m) * 64 + ncol] = acc.e[rr] + bias;
    }
}

// ---------------------------------------------------------------------------
// K3: per-anchor fused decode
// ---------------------------------------------------------------------------
__global__ void k_decode(const float* __restrict__ ho,
                         const int* __restrict__ imgh,
                         const int* __restrict__ imgw,
                         float* __restrict__ out,
                         float* __restrict__ boxes,
                         float* __restrict__ scores) {
    const int i = blockIdx.x * blockDim.x + threadIdx.x;
    if (i >= NANCH) return;
    const int p = i / 9, a = i - p * 9;
    const int hr = p >> 7, wc = p & 127;

    const float* hp = ho + (size_t)p * 64;
    const float l0 = hp[a * 4 + 0], l1 = hp[a * 4 + 1];
    const float l2 = hp[a * 4 + 2], l3 = hp[a * 4 + 3];
    const float c0 = hp[36 + a * 2], c1 = hp[36 + a * 2 + 1];

    out[R0_LOCS + (size_t)i * 4 + 0] = l0;
    out[R0_LOCS + (size_t)i * 4 + 1] = l1;
    out[R0_LOCS + (size_t)i * 4 + 2] = l2;
    out[R0_LOCS + (size_t)i * 4 + 3] = l3;
    out[R1_CLS + (size_t)i * 2 + 0] = c0;
    out[R1_CLS + (size_t)i * 2 + 1] = c1;

    const int ri = a / 3, si = a - ri * 3;
    const float scale = (si == 0) ? 8.0f : ((si == 1) ? 16.0f : 32.0f);
    const float sqr   = (ri == 0) ? 0.70710678118f : ((ri == 1) ? 1.0f : 1.41421356237f);
    const float ha = 16.0f * scale * sqr;
    const float wa = 16.0f * scale / sqr;
    const float cy = hr * 16.0f + 8.0f, cx = wc * 16.0f + 8.0f;
    out[R4_ANCH + (size_t)i * 4 + 0] = cy - 0.5f * ha;
    out[R4_ANCH + (size_t)i * 4 + 1] = cx - 0.5f * wa;
    out[R4_ANCH + (size_t)i * 4 + 2] = cy + 0.5f * ha;
    out[R4_ANCH + (size_t)i * 4 + 3] = cx + 0.5f * wa;

    const float ncy = cy + l0 * ha;
    const float ncx = cx + l1 * wa;
    const float nh = expf(l2) * ha;
    const float nw = expf(l3) * wa;
    const float IH = (float)imgh[0], IW = (float)imgw[0];
    const float y1 = fminf(fmaxf(ncy - 0.5f * nh, 0.0f), IH);
    const float x1 = fminf(fmaxf(ncx - 0.5f * nw, 0.0f), IW);
    const float y2 = fminf(fmaxf(ncy + 0.5f * nh, 0.0f), IH);
    const float x2 = fminf(fmaxf(ncx + 0.5f * nw, 0.0f), IW);
    boxes[(size_t)i * 4 + 0] = y1;
    boxes[(size_t)i * 4 + 1] = x1;
    boxes[(size_t)i * 4 + 2] = y2;
    boxes[(size_t)i * 4 + 3] = x2;

    const bool valid = ((y2 - y1) >= 16.0f) && ((x2 - x1) >= 16.0f);
    const float obj = 1.0f / (1.0f + expf(c0 - c1));
    scores[i] = valid ? obj : -__builtin_inff();
}

// ---------------------------------------------------------------------------
// K4: single-WG 4-pass radix select for the top-NUM_PRE set
// ---------------------------------------------------------------------------
__global__ __launch_bounds__(1024) void k_select(const float* __restrict__ scores,
                                                 unsigned* __restrict__ ctrl) {
    __shared__ int bins[256];
    __shared__ unsigned sprefix;
    __shared__ int sneed;
    const int tid = threadIdx.x;
    if (tid == 0) { sprefix = 0u; sneed = NUM_PRE; }
    __syncthreads();

    for (int pass = 0; pass < 4; ++pass) {
        const int shift = 24 - 8 * pass;
        if (tid < 256) bins[tid] = 0;
        __syncthreads();
        const unsigned prefix = sprefix;
        for (int i = tid; i < NANCH; i += 1024) {
            const unsigned u = fkey(scores[i]);
            if (pass == 0 || (u >> (shift + 8)) == prefix)
                atomicAdd(&bins[(u >> shift) & 255u], 1);
        }
        __syncthreads();
        if (tid == 0) {
            int acc = 0, nd = sneed, chosen = 0;
            for (int v = 255; v >= 0; --v) {
                if (acc + bins[v] >= nd) { chosen = v; break; }
                acc += bins[v];
            }
            sneed = nd - acc;
            sprefix = (sprefix << 8) | (unsigned)chosen;
        }
        __syncthreads();
    }
    if (tid == 0) {
        ctrl[0] = sprefix;
        ctrl[1] = (unsigned)sneed;
        ctrl[2] = 0u;
        ctrl[3] = 0u;
    }
}

// ---------------------------------------------------------------------------
// K5: compact top-NUM_PRE boxes/scores
// ---------------------------------------------------------------------------
__global__ void k_compact(const float* __restrict__ scores,
                          const float* __restrict__ boxes,
                          unsigned* __restrict__ ctrl,
                          float* __restrict__ topS,
                          float* __restrict__ topB) {
    const unsigned T = ctrl[0];
    const unsigned take_eq = ctrl[1];
    const unsigned base_eq = (unsigned)NUM_PRE - take_eq;
    const int stride = gridDim.x * blockDim.x;
    for (int i = blockIdx.x * blockDim.x + threadIdx.x; i < NANCH; i += stride) {
        const unsigned u = fkey(scores[i]);
        int pos = -1;
        if (u > T) {
            pos = (int)atomicAdd(&ctrl[2], 1u);
        } else if (u == T) {
            const unsigned e = atomicAdd(&ctrl[3], 1u);
            if (e < take_eq) pos = (int)(base_eq + e);
        }
        if (pos >= 0) {
            topS[pos] = scores[i];
            topB[(size_t)pos * 4 + 0] = boxes[(size_t)i * 4 + 0];
            topB[(size_t)pos * 4 + 1] = boxes[(size_t)i * 4 + 1];
            topB[(size_t)pos * 4 + 2] = boxes[(size_t)i * 4 + 2];
            topB[(size_t)pos * 4 + 3] = boxes[(size_t)i * 4 + 3];
        }
    }
}

// ---------------------------------------------------------------------------
// K6: greedy NMS, single WG, scores in LDS
// ---------------------------------------------------------------------------
__global__ __launch_bounds__(1024) void k_nms(const float* __restrict__ topB,
                                              const float* __restrict__ topS,
                                              int* __restrict__ keep) {
    __shared__ float sc[NUM_PRE];
    __shared__ float rv[1024];
    __shared__ int   ri[1024];
    __shared__ float bbx[4];
    __shared__ float barea;
    __shared__ int   bidx;
    const int tid = threadIdx.x;
    const float NEG = -__builtin_inff();

    for (int i = tid; i < NUM_PRE; i += 1024) sc[i] = topS[i];
    __syncthreads();

    for (int it = 0; it < NUM_POST; ++it) {
        float bv = NEG; int bi = -1;
        for (int i = tid; i < NUM_PRE; i += 1024) {
            const float s = sc[i];
            if (s > bv) { bv = s; bi = i; }
        }
        rv[tid] = bv; ri[tid] = bi;
        __syncthreads();
        for (int s = 512; s > 0; s >>= 1) {
            if (tid < s && rv[tid + s] > rv[tid]) { rv[tid] = rv[tid + s]; ri[tid] = ri[tid + s]; }
            __syncthreads();
        }
        if (tid == 0) {
            const int b = ri[0];
            bidx = b;
            keep[it] = b;
            if (b >= 0) {
                const float y1 = topB[b * 4 + 0], x1 = topB[b * 4 + 1];
                const float y2 = topB[b * 4 + 2], x2 = topB[b * 4 + 3];
                bbx[0] = y1; bbx[1] = x1; bbx[2] = y2; bbx[3] = x2;
                barea = (y2 - y1) * (x2 - x1);
                sc[b] = NEG;
            }
        }
        __syncthreads();
        const int b = bidx;
        if (b >= 0) {
            const float by1 = bbx[0], bx1 = bbx[1], by2 = bbx[2], bx2 = bbx[3], ba = barea;
            for (int i = tid; i < NUM_PRE; i += 1024) {
                if (sc[i] > NEG) {
                    const float y1 = topB[i * 4 + 0], x1 = topB[i * 4 + 1];
                    const float y2 = topB[i * 4 + 2], x2 = topB[i * 4 + 3];
                    const float yy1 = fmaxf(y1, by1), xx1 = fmaxf(x1, bx1);
                    const float yy2 = fminf(y2, by2), xx2 = fminf(x2, bx2);
                    const float inter = fmaxf(yy2 - yy1, 0.0f) * fmaxf(xx2 - xx1, 0.0f);
                    const float area = (y2 - y1) * (x2 - x1);
                    const float iou = inter / (area + ba - inter + 1e-9f);
                    if (iou > 0.7f) sc[i] = NEG;
                }
            }
        }
        __syncthreads();
    }
}

// ---------------------------------------------------------------------------
// K7: finalize padded outputs
// ---------------------------------------------------------------------------
__global__ void k_final(const float* __restrict__ topB,
                        const int* __restrict__ keep,
                        float* __restrict__ out) {
    const int it = blockIdx.x * blockDim.x + threadIdx.x;
    if (it >= NUM_POST) return;
    const int k = keep[it];
    float* pb = out + R2_BOXES;
    int*   pi = (int*)out + R3_IDX;
    float* pm = out + R5_MASK;
    if (k >= 0) {
        pb[it * 4 + 0] = topB[k * 4 + 0];
        pb[it * 4 + 1] = topB[k * 4 + 1];
        pb[it * 4 + 2] = topB[k * 4 + 2];
        pb[it * 4 + 3] = topB[k * 4 + 3];
        pi[it] = 0;
        pm[it] = 1.0f;
    } else {
        pb[it * 4 + 0] = 0.0f; pb[it * 4 + 1] = 0.0f;
        pb[it * 4 + 2] = 0.0f; pb[it * 4 + 3] = 0.0f;
        pi[it] = -1;
        pm[it] = 0.0f;
    }
}

// ---------------------------------------------------------------------------
// Host launcher
// ---------------------------------------------------------------------------
extern "C" void kernel_launch(void* const* d_in, const int* in_sizes, int n_in,
                              void* d_out, int out_size, void* d_ws, size_t ws_size,
                              hipStream_t stream) {
    (void)in_sizes; (void)n_in; (void)out_size; (void)ws_size;
    const float* x      = (const float*)d_in[0];
    const float* conv_w = (const float*)d_in[1];
    const float* conv_b = (const float*)d_in[2];
    const float* reg_w  = (const float*)d_in[3];
    const float* reg_b  = (const float*)d_in[4];
    const float* cls_w  = (const float*)d_in[5];
    const float* cls_b  = (const float*)d_in[6];
    const int*   img_h  = (const int*)d_in[7];
    const int*   img_w  = (const int*)d_in[8];
    float* out = (float*)d_out;

    char* ws = (char*)d_ws;
    h16_t*    xt    = (h16_t*)(ws + OFF_XT);
    h16_t*    whf   = (h16_t*)(ws + OFF_WF);
    h16_t*    hwf   = (h16_t*)(ws + OFF_HF);
    h16_t*    hh    = (h16_t*)(ws + OFF_HH);
    float*    ho    = (float*)(ws + OFF_HO);
    float*    boxes = (float*)(ws + OFF_BX);
    float*    score = (float*)(ws + OFF_SC);
    unsigned* ctrl  = (unsigned*)(ws + OFF_CT);
    float*    topS  = (float*)(ws + OFF_TS);
    float*    topB  = (float*)(ws + OFF_TB);
    int*      keep  = (int*)(ws + OFF_KP);

    k_convert<<<4096, 256, 0, stream>>>(x, conv_w, reg_w, cls_w, xt, whf, hwf);
    k_conv<<<dim3(HDIM, COUT / 128), 256, 0, stream>>>(xt, whf, conv_b, hh);
    k_head<<<NPOS / 16, 128, 0, stream>>>(hh, hwf, reg_b, cls_b, ho);
    k_decode<<<(NANCH + 255) / 256, 256, 0, stream>>>(ho, img_h, img_w, out, boxes, score);
    k_select<<<1, 1024, 0, stream>>>(score, ctrl);
    k_compact<<<576, 256, 0, stream>>>(score, boxes, ctrl, topS, topB);
    k_nms<<<1, 1024, 0, stream>>>(topB, topS, keep);
    k_final<<<(NUM_POST + 255) / 256, 256, 0, stream>>>(topB, keep, out);
}
